// MultiheadAttention_73581379715377
// MI455X (gfx1250) — matrix operand order repaired and checked by reading, not compiled, not run on hardware
//
#include <hip/hip_runtime.h>

#define L_SEQ 2048
#define DMODEL 1024
#define NH 16
#define DH 64
#define BATCH 2
#define MROWS (BATCH * L_SEQ)   // 4096
#define KSTEP 64                // keys per attention step
#define NTILES (L_SEQ / KSTEP)  // 32

typedef _Float16 v16h __attribute__((ext_vector_type(16)));
typedef _Float16 v8h  __attribute__((ext_vector_type(8)));
typedef float    v8f  __attribute__((ext_vector_type(8)));
typedef float    v4f  __attribute__((ext_vector_type(4)));
typedef unsigned int u32x4 __attribute__((ext_vector_type(4)));
typedef int          i32x4 __attribute__((ext_vector_type(4)));
typedef int          i32x8 __attribute__((ext_vector_type(8)));

#if defined(__has_builtin)
#if __has_builtin(__builtin_amdgcn_tensor_load_to_lds) && \
    __has_builtin(__builtin_amdgcn_s_wait_tensorcnt)
#define USE_TDM 1
#endif
#endif
#ifndef USE_TDM
#define USE_TDM 0
#endif

union F16x16 { v16h v; v8h h[2]; };

// Load 16 halves as two contiguous 8-half (16B) chunks -> one v16h fragment.
__device__ __forceinline__ v16h ld_frag(const _Float16* p0, const _Float16* p1) {
  F16x16 u;
  u.h[0] = *(const v8h*)p0;
  u.h[1] = *(const v8h*)p1;
  return u.v;
}

__device__ __forceinline__ v8f wmma_f16(v16h a, v16h b, v8f c) {
  return __builtin_amdgcn_wmma_f32_16x16x32_f16(
      /*neg_a=*/false, a, /*neg_b=*/false, b,
      /*c_mod=*/(short)0, c, /*reuse_a=*/false, /*reuse_b=*/false);
}

#if USE_TDM
// Issue one TDM 2-D tile load: tile_rows rows of tile_elems_x 4-byte elements,
// global row stride row_stride_elems (4-byte units), into LDS at lds_addr.
// D# layout per CDNA5 ISA ch.8 (group0 128b, group1 256b, groups 2/3 zero).
__device__ __forceinline__ void tdm_load_2d(unsigned lds_addr,
                                            const void* gaddr,
                                            unsigned tile_elems_x,
                                            unsigned tile_rows,
                                            unsigned row_stride_elems) {
  const unsigned long long ga = (unsigned long long)(size_t)gaddr;
  u32x4 g0;
  g0.x = 1u;                                       // count=1, user descriptor
  g0.y = lds_addr;                                 // LDS byte address
  g0.z = (unsigned)(ga & 0xffffffffu);             // global_addr[31:0]
  g0.w = (unsigned)((ga >> 32) & 0x01ffffffu)      // global_addr[56:32]
         | 0x80000000u;                            // type = 2 ("image")
  const unsigned data_size = 2u;                   // 4-byte elements
  const unsigned td0 = 1u << 24;                   // huge tensor dims:
  const unsigned td1 = 1u << 24;                   // OOB clamp never triggers
  i32x8 g1;
  g1[0] = (int)(data_size << 16);                  // mask=0, ds, no pad/iter
  g1[1] = (int)((td0 & 0xffffu) << 16);            // tensor_dim0[15:0]
  g1[2] = (int)((td0 >> 16) | ((td1 & 0xffffu) << 16));
  g1[3] = (int)((td1 >> 16) | (tile_elems_x << 16)); // tile_dim0
  g1[4] = (int)tile_rows;                          // tile_dim1 (tile_dim2=0)
  g1[5] = (int)row_stride_elems;                   // tensor_dim0_stride[31:0]
  g1[6] = 0;
  g1[7] = 0;
  const i32x4 z4 = {0, 0, 0, 0};
#if __clang_major__ >= 23
  const i32x8 z8 = {0, 0, 0, 0, 0, 0, 0, 0};
  __builtin_amdgcn_tensor_load_to_lds(g0, g1, z4, z4, z8, 0);
#else
  __builtin_amdgcn_tensor_load_to_lds(g0, g1, z4, z4, 0);
#endif
}
#endif  // USE_TDM

// ---------------------------------------------------------------------------
// Kernel 1: QKV projections. Y = X @ W^T, f32 in, f16 head-major out.
// blockIdx.z: 0 = Q -> [b,h,L,dh], 1 = K -> [b,h,L,dh], 2 = V -> [b,h,dh,L]
// ---------------------------------------------------------------------------
__global__ __launch_bounds__(256)
void mha_proj_qkv(const float* __restrict__ q, const float* __restrict__ k,
                  const float* __restrict__ v, const float* __restrict__ Wq,
                  const float* __restrict__ Wk, const float* __restrict__ Wv,
                  _Float16* __restrict__ Qh, _Float16* __restrict__ Kh,
                  _Float16* __restrict__ Vt) {
  __shared__ _Float16 sW[64 * 32];   // 4 KiB f16 weight chunk [e][d]

  const int mode = blockIdx.z;
  const float* X = (mode == 0) ? q : (mode == 1) ? k : v;
  const float* W = (mode == 0) ? Wq : (mode == 1) ? Wk : Wv;
  _Float16* out  = (mode == 0) ? Qh : (mode == 1) ? Kh : Vt;

  const int tid  = threadIdx.x;
  const int lane = tid & 31;
  const int wave = tid >> 5;
  const int m0   = blockIdx.x * 128 + wave * 16;
  const int e0   = blockIdx.y * 64;
  const int half = (lane >> 4) & 1;
  const int l15  = lane & 15;
  const int srow = tid >> 2;
  const int scol = (tid & 3) * 8;

  v8f acc[4];
#pragma unroll
  for (int t = 0; t < 4; ++t) acc[t] = v8f{};

  const float* xr = X + (size_t)(m0 + l15) * DMODEL;

  for (int d0 = 0; d0 < DMODEL; d0 += 32) {
    // ---- stage W[e0:e0+64][d0:d0+32] -> f16 LDS ----
    {
      const float* wg = W + (size_t)(e0 + srow) * DMODEL + d0 + scol;
      v4f w0 = *(const v4f*)wg;
      v4f w1 = *(const v4f*)(wg + 4);
      v8h hv;
#pragma unroll
      for (int j = 0; j < 4; ++j) {
        hv[j]     = (_Float16)w0[j];
        hv[4 + j] = (_Float16)w1[j];
      }
      *(v8h*)(sW + srow * 32 + scol) = hv;
    }
    __syncthreads();

    // ---- A fragment (16x32 of X), f32 -> f16 inline ----
    const int c0 = d0 + half * 8;
    v4f x0 = *(const v4f*)(xr + c0);
    v4f x1 = *(const v4f*)(xr + c0 + 4);
    v4f x2 = *(const v4f*)(xr + c0 + 16);
    v4f x3 = *(const v4f*)(xr + c0 + 20);
    __builtin_prefetch(xr + c0 + 32, 0, 1);   // speculative: safe past end
    v16h av;
#pragma unroll
    for (int j = 0; j < 4; ++j) {
      av[j]      = (_Float16)x0[j];
      av[4 + j]  = (_Float16)x1[j];
      av[8 + j]  = (_Float16)x2[j];
      av[12 + j] = (_Float16)x3[j];
    }
    // ---- B fragments from LDS ----
#pragma unroll
    for (int t = 0; t < 4; ++t) {
      const _Float16* wr = sW + (t * 16 + l15) * 32 + half * 8;
      v16h bv = ld_frag(wr, wr + 16);
      acc[t] = wmma_f16(av, bv, acc[t]);
    }
    __syncthreads();
  }

  // ---- write-out in head-major f16 layouts ----
#pragma unroll
  for (int t = 0; t < 4; ++t) {
#pragma unroll
    for (int r = 0; r < 8; ++r) {
      const int gm = m0 + r + half * 8;
      const int e  = e0 + t * 16 + l15;
      const int b  = gm >> 11;
      const int l  = gm & (L_SEQ - 1);
      const int hh = e >> 6;
      const int dd = e & (DH - 1);
      const _Float16 val = (_Float16)acc[t][r];
      if (mode < 2)
        out[(((size_t)(b * NH + hh)) * L_SEQ + l) * DH + dd] = val;
      else
        out[(((size_t)(b * NH + hh)) * DH + dd) * L_SEQ + l] = val;
    }
  }
}

// ---------------------------------------------------------------------------
// Kernel 2: fused flash attention.
// Block = 4 waves covering one 64-query block of one (b,h). K/V tiles are
// double-buffered in LDS and streamed by the Tensor Data Mover one step
// ahead of compute (TENSORcnt pipelining); falls back to cooperative b128
// copies when the TDM builtin is unavailable.
// ---------------------------------------------------------------------------
__global__ __launch_bounds__(128)
void mha_flash_attn(const _Float16* __restrict__ Qh,
                    const _Float16* __restrict__ Kh,
                    const _Float16* __restrict__ Vt,
                    const int* __restrict__ kpm,
                    const float* __restrict__ amask,
                    _Float16* __restrict__ ctx) {
  __shared__ _Float16 sK[2][KSTEP * DH];     // [key][dh]   2 x 8 KiB
  __shared__ _Float16 sV[2][DH * KSTEP];     // [dh][key]   2 x 8 KiB
  __shared__ _Float16 sP[4][16 * KSTEP];     // per-wave P  8 KiB

  const int tid  = threadIdx.x;
  const int lane = tid & 31;
  const int wave = tid >> 5;
  const int nqb  = L_SEQ / 64;               // 32 query blocks per (b,h)
  const int bh   = blockIdx.x / nqb;
  const int q0   = (blockIdx.x % nqb) * 64 + wave * 16;
  const int b    = bh >> 4;
  const int h    = bh & 15;
  const int half = (lane >> 4) & 1;
  const int l15  = lane & 15;

  const _Float16* Qp = Qh + (size_t)bh * L_SEQ * DH;
  const _Float16* Kp = Kh + (size_t)bh * L_SEQ * DH;
  const _Float16* Vp = Vt + (size_t)bh * DH * L_SEQ;

  // Q A-fragments for dh chunks 0-31 / 32-63 held in registers.
  const _Float16* qr = Qp + (size_t)(q0 + l15) * DH;
  const v16h a0 = ld_frag(qr + half * 8,      qr + half * 8 + 16);
  const v16h a1 = ld_frag(qr + 32 + half * 8, qr + 32 + half * 8 + 16);

  float mrun[8], lrun[8];
  v8f acc[4];
#pragma unroll
  for (int r = 0; r < 8; ++r) { mrun[r] = -3.0e38f; lrun[r] = 0.0f; }
#pragma unroll
  for (int t = 0; t < 4; ++t) acc[t] = v8f{};

  const float scale = 0.03125f;  // 1/sqrt(1024)

#if USE_TDM
  // Prologue: start streaming tile 0 into buffer 0.
  if (wave == 0) {
    tdm_load_2d((unsigned)(size_t)(void*)sK[0], Kp,
                KSTEP * DH / 2, 1, KSTEP * DH / 2);
    tdm_load_2d((unsigned)(size_t)(void*)sV[0], Vp,
                KSTEP / 2, DH, L_SEQ / 2);
  }
#endif

  for (int i = 0; i < NTILES; ++i) {
    const int k0  = i * KSTEP;
    const int buf = i & 1;
#if USE_TDM
    if (wave == 0) {
      if (i + 1 < NTILES) {
        // Stream the next tile into the other buffer (sealed by the
        // previous iteration's closing barrier), then allow exactly those
        // two newest descriptors to stay in flight: in-order TDM completion
        // guarantees tile i is resident.
        const int nbuf = buf ^ 1;
        tdm_load_2d((unsigned)(size_t)(void*)sK[nbuf],
                    Kp + (size_t)(k0 + KSTEP) * DH,
                    KSTEP * DH / 2, 1, KSTEP * DH / 2);
        tdm_load_2d((unsigned)(size_t)(void*)sV[nbuf],
                    Vp + k0 + KSTEP, KSTEP / 2, DH, L_SEQ / 2);
        __builtin_amdgcn_s_wait_tensorcnt(2);
      } else {
        __builtin_amdgcn_s_wait_tensorcnt(0);
      }
    }
#else
    {
      const v8h* gk = (const v8h*)(Kp + (size_t)k0 * DH);
      v8h* lk = (v8h*)sK[buf];
#pragma unroll
      for (int c = 0; c < 4; ++c) lk[tid + c * 128] = gk[tid + c * 128];
#pragma unroll
      for (int c = 0; c < 4; ++c) {
        const int chunk = tid + c * 128;        // 0..511
        const int row   = chunk >> 3;
        const int col8  = (chunk & 7) * 8;
        *(v8h*)(sV[buf] + row * KSTEP + col8) =
            *(const v8h*)(Vp + (size_t)row * L_SEQ + k0 + col8);
      }
    }
#endif
    __syncthreads();

    // ---- scores: four 16-key groups ----
    v8f s[4];
#pragma unroll
    for (int g = 0; g < 4; ++g) {
      const _Float16* kr = sK[buf] + (g * 16 + l15) * DH;
      v16h b0 = ld_frag(kr + half * 8,      kr + half * 8 + 16);
      v16h b1 = ld_frag(kr + 32 + half * 8, kr + 32 + half * 8 + 16);
      v8f z{};
      z = wmma_f16(a0, b0, z);
      z = wmma_f16(a1, b1, z);
      s[g] = z;
    }
    // ---- scale + attn_mask add + key-padding replacement (-1e7) ----
#pragma unroll
    for (int g = 0; g < 4; ++g) {
      const int kcol = k0 + g * 16 + l15;
      const bool pad = (kpm[b * L_SEQ + kcol] == 0);
#pragma unroll
      for (int r = 0; r < 8; ++r) {
        const int qrow = q0 + r + half * 8;
        float sv = s[g][r] * scale + amask[(size_t)qrow * L_SEQ + kcol];
        s[g][r] = pad ? -1.0e7f : sv;
      }
    }
    // ---- online softmax (rows r / r+8 across lane halves) ----
    _Float16* lp = sP[wave];
#pragma unroll
    for (int r = 0; r < 8; ++r) {
      float x0 = s[0][r], x1 = s[1][r], x2 = s[2][r], x3 = s[3][r];
      float mx = fmaxf(fmaxf(x0, x1), fmaxf(x2, x3));
#pragma unroll
      for (int off = 8; off; off >>= 1) mx = fmaxf(mx, __shfl_xor(mx, off, 16));
      const float mnew  = fmaxf(mrun[r], mx);
      const float alpha = __expf(mrun[r] - mnew);
      mrun[r] = mnew;
      const float p0 = __expf(x0 - mnew);
      const float p1 = __expf(x1 - mnew);
      const float p2 = __expf(x2 - mnew);
      const float p3 = __expf(x3 - mnew);
      float ps = (p0 + p1) + (p2 + p3);
#pragma unroll
      for (int off = 8; off; off >>= 1) ps += __shfl_xor(ps, off, 16);
      lrun[r] = lrun[r] * alpha + ps;
#pragma unroll
      for (int t = 0; t < 4; ++t) acc[t][r] *= alpha;
      const int row = r + half * 8;
      lp[row * KSTEP + l15]      = (_Float16)p0;
      lp[row * KSTEP + 16 + l15] = (_Float16)p1;
      lp[row * KSTEP + 32 + l15] = (_Float16)p2;
      lp[row * KSTEP + 48 + l15] = (_Float16)p3;
    }
    asm volatile("s_wait_dscnt 0" ::: "memory");
    // ---- reload P as two 16x32 A-fragments ----
    const _Float16* pr = lp + l15 * KSTEP + half * 8;
    const v16h pa0 = ld_frag(pr,      pr + 16);
    const v16h pa1 = ld_frag(pr + 32, pr + 48);
    // ---- O += P x V ----
#pragma unroll
    for (int t = 0; t < 4; ++t) {
      const _Float16* vr = sV[buf] + (t * 16 + l15) * KSTEP;
      v16h bv0 = ld_frag(vr + half * 8,      vr + half * 8 + 16);
      v16h bv1 = ld_frag(vr + 32 + half * 8, vr + 32 + half * 8 + 16);
      acc[t] = wmma_f16(pa0, bv0, acc[t]);
      acc[t] = wmma_f16(pa1, bv1, acc[t]);
    }
    __syncthreads();
  }

  // ---- normalize and write context [B*L, d] as f16 ----
#pragma unroll
  for (int t = 0; t < 4; ++t) {
#pragma unroll
    for (int r = 0; r < 8; ++r) {
      const int qrow = q0 + r + half * 8;
      const float o = acc[t][r] / lrun[r];
      ctx[((size_t)(b * L_SEQ + qrow)) * DMODEL + h * DH + t * 16 + l15] =
          (_Float16)o;
    }
  }
}

// ---------------------------------------------------------------------------
// Kernel 3: output projection. out = ctx @ Wo^T, f32 result.
// ---------------------------------------------------------------------------
__global__ __launch_bounds__(256)
void mha_out_proj(const _Float16* __restrict__ ctx,
                  const float* __restrict__ Wo,
                  float* __restrict__ out) {
  __shared__ _Float16 sW[64 * 32];

  const int tid  = threadIdx.x;
  const int lane = tid & 31;
  const int wave = tid >> 5;
  const int m0   = blockIdx.x * 128 + wave * 16;
  const int e0   = blockIdx.y * 64;
  const int half = (lane >> 4) & 1;
  const int l15  = lane & 15;
  const int srow = tid >> 2;
  const int scol = (tid & 3) * 8;

  v8f acc[4];
#pragma unroll
  for (int t = 0; t < 4; ++t) acc[t] = v8f{};

  const _Float16* xr = ctx + (size_t)(m0 + l15) * DMODEL;

  for (int d0 = 0; d0 < DMODEL; d0 += 32) {
    {
      const float* wg = Wo + (size_t)(e0 + srow) * DMODEL + d0 + scol;
      v4f w0 = *(const v4f*)wg;
      v4f w1 = *(const v4f*)(wg + 4);
      v8h hv;
#pragma unroll
      for (int j = 0; j < 4; ++j) {
        hv[j]     = (_Float16)w0[j];
        hv[4 + j] = (_Float16)w1[j];
      }
      *(v8h*)(sW + srow * 32 + scol) = hv;
    }
    __syncthreads();

    const int c0 = d0 + half * 8;
    v16h av = ld_frag(xr + c0, xr + c0 + 16);
    __builtin_prefetch(xr + c0 + 32, 0, 1);
#pragma unroll
    for (int t = 0; t < 4; ++t) {
      const _Float16* wr = sW + (t * 16 + l15) * 32 + half * 8;
      v16h bv = ld_frag(wr, wr + 16);
      acc[t] = wmma_f16(av, bv, acc[t]);
    }
    __syncthreads();
  }

#pragma unroll
  for (int t = 0; t < 4; ++t) {
#pragma unroll
    for (int r = 0; r < 8; ++r) {
      const int gm = m0 + r + half * 8;
      out[(size_t)gm * DMODEL + e0 + t * 16 + l15] = acc[t][r];
    }
  }
}

// ---------------------------------------------------------------------------
extern "C" void kernel_launch(void* const* d_in, const int* in_sizes, int n_in,
                              void* d_out, int out_size, void* d_ws,
                              size_t ws_size, hipStream_t stream) {
  (void)in_sizes; (void)n_in; (void)out_size; (void)ws_size;
  const float* q     = (const float*)d_in[0];
  const float* k     = (const float*)d_in[1];
  const float* v     = (const float*)d_in[2];
  const int*   kpm   = (const int*)d_in[3];
  const float* amask = (const float*)d_in[4];
  const float* Wq    = (const float*)d_in[5];
  const float* Wk    = (const float*)d_in[6];
  const float* Wv    = (const float*)d_in[7];
  const float* Wo    = (const float*)d_in[8];

  // Workspace layout (f16): Qh | Kh | Vt | ctx, 8 MiB each = 32 MiB total.
  _Float16* Qh  = (_Float16*)d_ws;
  _Float16* Kh  = Qh + (size_t)MROWS * DMODEL;
  _Float16* Vt  = Kh + (size_t)MROWS * DMODEL;
  _Float16* ctx = Vt + (size_t)MROWS * DMODEL;

  mha_proj_qkv<<<dim3(32, 16, 3), 256, 0, stream>>>(q, k, v, Wq, Wk, Wv,
                                                    Qh, Kh, Vt);
  mha_flash_attn<<<dim3(1024), 128, 0, stream>>>(Qh, Kh, Vt, kpm, amask, ctx);
  mha_out_proj<<<dim3(32, 16), 256, 0, stream>>>(ctx, Wo, (float*)d_out);
}
